// PatternFlowBranch_17171279249960
// MI455X (gfx1250) — matrix operand order repaired and checked by reading, not compiled, not run on hardware
//
#include <hip/hip_runtime.h>
#include <cstdint>

// ---------------------------------------------------------------------------
// MI455X (gfx1250) transformer-encoder forward.
// - all matmuls via v_wmma_f32_16x16x32_bf16 (fp32 accumulate)
// - LDS tiles filled with global_load_async_to_lds_b128 (ASYNCcnt pipelined)
// - flash attention with transposed-V LDS staging (packed b32 stores,
//   vector 32B DS fragment reads)
// - cross-lane reductions via v_permlane16_b32 (VALU, no LDS bpermute)
// ---------------------------------------------------------------------------

typedef __bf16 bf16_t;
typedef __attribute__((ext_vector_type(16))) __bf16 v16bf;
typedef __attribute__((ext_vector_type(8)))  __bf16 v8bf;
typedef __attribute__((ext_vector_type(8)))  float  v8f;

constexpr int N_TOK = 2048;
constexpr int D_IN  = 512;
constexpr int HDIM  = 1024;   // H
constexpr int D_OUT = 256;
constexpr int NHEAD = 16;
constexpr int HD    = 64;     // head dim
constexpr int NLAY  = 4;

__device__ __forceinline__ v8f wmma_bf16(v16bf a, v16bf b, v8f c) {
  // 8 args: (neg_a, A, neg_b, B, c_mod, C, reuse_a, reuse_b)
  return __builtin_amdgcn_wmma_f32_16x16x32_bf16(false, a, false, b, (short)0, c, false, false);
}

// ---------------------------------------------------------------------------
// VALU cross-lane butterflies (v_permlane16_b32 / v_permlanex16_b32).
// Selector nibbles: result lane j (within its 16-lane row) takes row element
// sel[j]; for a xor-m butterfly sel[j] = j ^ m.
// ---------------------------------------------------------------------------
#if __has_builtin(__builtin_amdgcn_permlane16)
#define XOR16_F(v, m, lo, hi)                                                   \
  __uint_as_float(__builtin_amdgcn_permlane16(                                  \
      __float_as_uint(v), __float_as_uint(v), (lo), (hi), false, false))
#else
#define XOR16_F(v, m, lo, hi) __shfl_xor((v), (m), 32)
#endif

__device__ __forceinline__ float xor1_f(float v)  { return XOR16_F(v, 1, 0x67452301u, 0xEFCDAB89u); }
__device__ __forceinline__ float xor2_f(float v)  { return XOR16_F(v, 2, 0x54761032u, 0xDCFE98BAu); }
__device__ __forceinline__ float xor4_f(float v)  { return XOR16_F(v, 4, 0x32107654u, 0xBA98FEDCu); }
__device__ __forceinline__ float xor8_f(float v)  { return XOR16_F(v, 8, 0xFEDCBA98u, 0x76543210u); }

__device__ __forceinline__ float xor16_f(float v) {
#if __has_builtin(__builtin_amdgcn_permlanex16)
  return __uint_as_float(__builtin_amdgcn_permlanex16(
      __float_as_uint(v), __float_as_uint(v), 0x76543210u, 0xFEDCBA98u, false, false));
#else
  return __shfl_xor(v, 16, 32);
#endif
}

// reduce within each 16-lane group; all 16 lanes end with the result
__device__ __forceinline__ float red16_max(float v) {
  v = fmaxf(v, xor1_f(v)); v = fmaxf(v, xor2_f(v));
  v = fmaxf(v, xor4_f(v)); v = fmaxf(v, xor8_f(v));
  return v;
}
__device__ __forceinline__ float red16_sum(float v) {
  v += xor1_f(v); v += xor2_f(v); v += xor4_f(v); v += xor8_f(v);
  return v;
}
// reduce across the full wave32
__device__ __forceinline__ float red32_sum(float v) {
  v += xor1_f(v); v += xor2_f(v); v += xor4_f(v); v += xor8_f(v); v += xor16_f(v);
  return v;
}

// Async global -> LDS copy of 16 bytes per lane (CDNA5 ASYNCcnt path).
__device__ __forceinline__ void async_copy_b128(uint32_t lds_byte_addr, const void* gptr) {
  asm volatile("global_load_async_to_lds_b128 %0, %1, off"
               :: "v"(lds_byte_addr), "v"((uint64_t)(uintptr_t)gptr)
               : "memory");
}
__device__ __forceinline__ void wait_async0() {
  asm volatile("s_wait_asynccnt 0x0" ::: "memory");
}
__device__ __forceinline__ void wait_async1() {
  asm volatile("s_wait_asynccnt 0x1" ::: "memory");
}
__device__ __forceinline__ uint32_t lds_addr_of(const void* p) {
  // LDS aperture: low 32 bits of the generic address are the LDS byte address.
  return (uint32_t)(uintptr_t)p;
}

// A fragment (16x32 MxK, bf16) from row-major [*, lda] matrix (global).
// lane<16: row=mBase+lane,     K = kBase   + {0..7, 16..23}
// lane>=16: row=mBase+lane-16, K = kBase+8 + {0..7, 16..23}
__device__ __forceinline__ v16bf load_a_frag(const bf16_t* __restrict__ A, int lda,
                                             int mBase, int kBase, int lane) {
  int row  = mBase + (lane & 15);
  int kOff = kBase + ((lane >> 4) ? 8 : 0);
  const v8bf* p = (const v8bf*)(A + (size_t)row * lda + kOff);
  v8bf lo = p[0];   // K = kOff .. kOff+7
  v8bf hi = p[2];   // K = kOff+16 .. kOff+23
  return __builtin_shufflevector(lo, hi, 0,1,2,3,4,5,6,7,8,9,10,11,12,13,14,15);
}

// B fragment (32x16 KxN) from an LDS tile laid out [n][K]: lane holds
// n = nRel + (lane&15); lane<16 reads K=kBase+{0..7,16..23},
// lane>=16 reads K=kBase+8+{0..7,16..23}.
__device__ __forceinline__ v16bf load_b_frag_lds(const bf16_t* t, int ldt,
                                                 int nRel, int kBase, int lane) {
  int col  = nRel + (lane & 15);
  int kOff = kBase + ((lane >> 4) ? 8 : 0);
  const v8bf* p = (const v8bf*)(t + col * ldt + kOff);
  v8bf lo = p[0];
  v8bf hi = p[2];
  return __builtin_shufflevector(lo, hi, 0,1,2,3,4,5,6,7,8,9,10,11,12,13,14,15);
}

// ---------------------------------------------------------------------------
// fp32 -> bf16 convert
// ---------------------------------------------------------------------------
__global__ void cvt_f32_bf16_kernel(const float* __restrict__ in,
                                    bf16_t* __restrict__ out, int n) {
  int i = blockIdx.x * blockDim.x + threadIdx.x;
  if (i < n) out[i] = (bf16_t)in[i];
}

// ---------------------------------------------------------------------------
// GEMM: out = act(A[M,K](bf16) @ W[Nn,K](bf16)^T + bias)   (row-major)
// grid = (Nn/64, M/128), block = 256 (8 waves); wave -> 16 rows x 64 cols.
// B tile (64 cols x 32 K = 4KB) staged in LDS by async copies, double
// buffered, issued one k-step ahead (ASYNCcnt in-order completion).
// ---------------------------------------------------------------------------
template <int RELU, int STORE_F32, int STORE_BF16>
__global__ __launch_bounds__(256) void gemm_bf16_kernel(
    const bf16_t* __restrict__ A, const bf16_t* __restrict__ W,
    const float* __restrict__ bias,
    float* __restrict__ outF, bf16_t* __restrict__ outB,
    int M, int Nn, int K) {
  __shared__ bf16_t btile[2][64 * 32];  // [colRel][k], 4KB per buffer

  int tid  = threadIdx.x;
  int wave = tid >> 5;
  int lane = tid & 31;
  int mBase = blockIdx.y * 128 + wave * 16;
  int nBase = blockIdx.x * 64;

  // one 16B async chunk per thread fills a 4KB tile
  auto issue_b = [&](int k, int buf) {
    int col = tid >> 2;       // 0..63
    int seg = tid & 3;        // 4 x 8 elements per 32-element row
    const bf16_t* g = W + (size_t)(nBase + col) * K + k + seg * 8;
    async_copy_b128(lds_addr_of(&btile[buf][col * 32 + seg * 8]), g);
  };

  const int KT = K >> 5;
  issue_b(0, 0);

  v8f acc[4] = {};
  for (int kt = 0; kt < KT; ++kt) {
    int k = kt << 5;
    if (kt + 1 < KT) { issue_b(k + 32, (kt + 1) & 1); wait_async1(); }
    else             { wait_async0(); }
    __syncthreads();

    const bf16_t* bt = btile[kt & 1];
    v16bf a = load_a_frag(A, K, mBase, k, lane);
#pragma unroll
    for (int nt = 0; nt < 4; ++nt) {
      v16bf b = load_b_frag_lds(bt, 32, nt * 16, 0, lane);
      acc[nt] = wmma_bf16(a, b, acc[nt]);
    }
    __syncthreads();
  }

  int hi = lane >> 4;
  int ln = lane & 15;
#pragma unroll
  for (int nt = 0; nt < 4; ++nt) {
#pragma unroll
    for (int j = 0; j < 8; ++j) {
      int row = mBase + j + 8 * hi;
      int col = nBase + nt * 16 + ln;
      float v = acc[nt][j] + bias[col];
      if (RELU) v = fmaxf(v, 0.0f);
      if (STORE_F32)  outF[(size_t)row * Nn + col] = v;
      if (STORE_BF16) outB[(size_t)row * Nn + col] = (bf16_t)v;
    }
  }
}

// ---------------------------------------------------------------------------
// Flash attention over one head / 128-query-row block.
// qkv: [N_TOK, 3H] bf16 (Q | K | V per row).  outB: [N_TOK, H] bf16.
// grid = (NHEAD, N_TOK/128), block = 256 (8 waves, 16 query rows each).
// K block async-copied to LDS; V block stored transposed (two adjacent key
// rows packed per 32-bit DS store) so every WMMA fragment read is one
// two aligned 16-byte DS loads. Double buffered, one barrier per key-block.
// Softmax reductions are permlane16 butterflies.
// ---------------------------------------------------------------------------
__global__ __launch_bounds__(256) void attn_kernel(
    const bf16_t* __restrict__ qkv, bf16_t* __restrict__ outB, float scale) {
  __shared__ bf16_t ktile[2][64 * 64];   // [key][hd]              8KB x2
  __shared__ bf16_t vtileT[2][64 * 64];  // [hd][key] (transposed) 8KB x2
  __shared__ bf16_t ptile[8][16 * 64];   // per-wave P staging     16KB

  const int ldq = 3 * HDIM;
  int tid  = threadIdx.x;
  int wave = tid >> 5;
  int lane = tid & 31;
  int hi   = lane >> 4;
  int ln   = lane & 15;
  int head  = blockIdx.x;
  int qBase = blockIdx.y * 128 + wave * 16;

  const bf16_t* Q  = qkv + head * HD;
  const bf16_t* Kp = qkv + HDIM + head * HD;
  const bf16_t* Vp = qkv + 2 * HDIM + head * HD;

  // Q fragments (head dim 64 -> 2 wmma K-steps), kept in registers.
  v16bf qf[2];
  qf[0] = load_a_frag(Q, ldq, qBase, 0, lane);
  qf[1] = load_a_frag(Q, ldq, qBase, 32, lane);

  float m[8], l[8];
#pragma unroll
  for (int j = 0; j < 8; ++j) { m[j] = -1e30f; l[j] = 0.0f; }
  v8f oacc[4] = {};

  for (int kb = 0, it = 0; kb < N_TOK; kb += 64, ++it) {
    int buf = it & 1;

    // prefetch next key block into L2 while we work on this one
    if (kb + 64 < N_TOK) {
      __builtin_prefetch(Kp + (size_t)(kb + 64) * ldq, 0, 0);
      __builtin_prefetch(Vp + (size_t)(kb + 64) * ldq, 0, 0);
    }

    // --- K block [64 keys x 64 hd]: async global->LDS, 2 chunks/thread ---
#pragma unroll
    for (int c = 0; c < 2; ++c) {
      int chunk = tid + c * 256;          // 0..511 (16B each)
      int r = chunk >> 3, seg = chunk & 7;
      const bf16_t* g = Kp + (size_t)(kb + r) * ldq + seg * 8;
      async_copy_b128(lds_addr_of(&ktile[buf][r * 64 + seg * 8]), g);
    }

    // --- V block stored transposed: read 8 cols of 2 adjacent key rows,
    //     pack the row pair into one 32-bit DS store per column ---
    {
      int rp = tid >> 3;             // 0..31 : key row pair (rows 2rp, 2rp+1)
      int c0 = (tid & 7) * 8;        // 0..56 : 8 hd columns
      const v8bf* g0 = (const v8bf*)(Vp + (size_t)(kb + 2 * rp) * ldq + c0);
      const v8bf* g1 = (const v8bf*)(Vp + (size_t)(kb + 2 * rp + 1) * ldq + c0);
      v8bf r0 = *g0, r1 = *g1;
      uint32_t* vt32 = (uint32_t*)&vtileT[buf][0];
#pragma unroll
      for (int e = 0; e < 8; ++e) {
        uint32_t lo = (uint32_t)__builtin_bit_cast(uint16_t, r0[e]);
        uint32_t hh = (uint32_t)__builtin_bit_cast(uint16_t, r1[e]);
        vt32[(c0 + e) * 32 + rp] = lo | (hh << 16);
      }
    }

    wait_async0();
    __syncthreads();

    // --- S = Q @ K^T (16 x 64 per wave), K fragments from LDS ---
    v8f s[4] = {};
#pragma unroll
    for (int nt = 0; nt < 4; ++nt) {
#pragma unroll
      for (int ks = 0; ks < 2; ++ks) {
        v16bf kf = load_b_frag_lds(&ktile[buf][0], 64, nt * 16, ks * 32, lane);
        s[nt] = wmma_bf16(qf[ks], kf, s[nt]);
      }
    }

    // fold the 1/sqrt(hd) scale into S once
#pragma unroll
    for (int nt = 0; nt < 4; ++nt)
#pragma unroll
      for (int j = 0; j < 8; ++j) s[nt][j] *= scale;

    // --- online softmax (rows live across 16-lane groups, VALU permlanes) ---
    float corr[8];
#pragma unroll
    for (int j = 0; j < 8; ++j) {
      float v = fmaxf(fmaxf(s[0][j], s[1][j]), fmaxf(s[2][j], s[3][j]));
      v = red16_max(v);
      float mn = fmaxf(m[j], v);
      corr[j] = __expf(m[j] - mn);
      m[j] = mn;
    }

    float rsum[8];
#pragma unroll
    for (int j = 0; j < 8; ++j) {
      float p0 = __expf(s[0][j] - m[j]);
      float p1 = __expf(s[1][j] - m[j]);
      float p2 = __expf(s[2][j] - m[j]);
      float p3 = __expf(s[3][j] - m[j]);
      s[0][j] = p0; s[1][j] = p1; s[2][j] = p2; s[3][j] = p3;
      rsum[j] = red16_sum((p0 + p1) + (p2 + p3));
    }
#pragma unroll
    for (int j = 0; j < 8; ++j) l[j] = l[j] * corr[j] + rsum[j];
#pragma unroll
    for (int nt = 0; nt < 4; ++nt)
#pragma unroll
      for (int j = 0; j < 8; ++j) oacc[nt][j] *= corr[j];

    // --- stage P (bf16) per-wave in LDS, then O += P @ V ---
    bf16_t* pt = &ptile[wave][0];
#pragma unroll
    for (int nt = 0; nt < 4; ++nt)
#pragma unroll
      for (int j = 0; j < 8; ++j)
        pt[(j + 8 * hi) * 64 + nt * 16 + ln] = (bf16_t)s[nt][j];

#pragma unroll
    for (int ks = 0; ks < 2; ++ks) {
      // A fragment of P from LDS (same-wave data; DS ops are in-order)
      v16bf pa;
      {
        const bf16_t* pp = pt + ln * 64 + ks * 32 + (hi ? 8 : 0);
        v8bf lo = *(const v8bf*)(pp);
        v8bf hh = *(const v8bf*)(pp + 16);
        pa = __builtin_shufflevector(lo, hh, 0,1,2,3,4,5,6,7,8,9,10,11,12,13,14,15);
      }
#pragma unroll
      for (int nt = 0; nt < 4; ++nt) {
        // B fragment of V from transposed LDS tile: two 16B reads at K and K+16
        v16bf vb = load_b_frag_lds(&vtileT[buf][0], 64, nt * 16, ks * 32, lane);
        oacc[nt] = wmma_bf16(pa, vb, oacc[nt]);
      }
    }
    // double buffer + the barrier above make one barrier per block safe
  }

  // --- normalize and store O ---
#pragma unroll
  for (int nt = 0; nt < 4; ++nt)
#pragma unroll
    for (int j = 0; j < 8; ++j) {
      int row = qBase + j + 8 * hi;
      int col = head * HD + nt * 16 + ln;
      outB[(size_t)row * HDIM + col] = (bf16_t)(oacc[nt][j] / l[j]);
    }
}

// ---------------------------------------------------------------------------
// h = LayerNorm(a + h) * g + b ; writes h (fp32, in place) and hB (bf16)
// one block (256 threads) per row of D=1024
// ---------------------------------------------------------------------------
__global__ __launch_bounds__(256) void add_ln_kernel(
    const float* __restrict__ a, float* __restrict__ h, bf16_t* __restrict__ hB,
    const float* __restrict__ g, const float* __restrict__ b) {
  const int D = HDIM;
  int row = blockIdx.x;
  const float* ar = a + (size_t)row * D;
  float* hr = h + (size_t)row * D;
  bf16_t* hbr = hB + (size_t)row * D;
  int wave = threadIdx.x >> 5, lane = threadIdx.x & 31;
  __shared__ float sb[8], sb2[8];

  float loc[4];
  float s = 0.0f;
#pragma unroll
  for (int i = 0; i < 4; ++i) {
    int c = threadIdx.x + i * 256;
    loc[i] = ar[c] + hr[c];
    s += loc[i];
  }
  s = red32_sum(s);
  if (lane == 0) sb[wave] = s;
  __syncthreads();
  float tot = 0.0f;
#pragma unroll
  for (int w = 0; w < 8; ++w) tot += sb[w];
  float mean = tot * (1.0f / D);

  float sq = 0.0f;
#pragma unroll
  for (int i = 0; i < 4; ++i) { float d = loc[i] - mean; sq += d * d; }
  sq = red32_sum(sq);
  if (lane == 0) sb2[wave] = sq;
  __syncthreads();
  float vtot = 0.0f;
#pragma unroll
  for (int w = 0; w < 8; ++w) vtot += sb2[w];
  float rstd = rsqrtf(vtot * (1.0f / D) + 1e-5f);

#pragma unroll
  for (int i = 0; i < 4; ++i) {
    int c = threadIdx.x + i * 256;
    float v = (loc[i] - mean) * rstd * g[c] + b[c];
    hr[c] = v;
    hbr[c] = (bf16_t)v;
  }
}

// ---------------------------------------------------------------------------
// out = LayerNorm(y) * g + b over D_OUT=256 ; one block per row, 1 elem/thread
// ---------------------------------------------------------------------------
__global__ __launch_bounds__(256) void final_ln_kernel(
    const float* __restrict__ y, const float* __restrict__ g,
    const float* __restrict__ b, float* __restrict__ out) {
  const int D = D_OUT;
  int row = blockIdx.x;
  int wave = threadIdx.x >> 5, lane = threadIdx.x & 31;
  __shared__ float sb[8], sb2[8];

  float v = y[(size_t)row * D + threadIdx.x];
  float s = red32_sum(v);
  if (lane == 0) sb[wave] = s;
  __syncthreads();
  float tot = 0.0f;
#pragma unroll
  for (int w = 0; w < 8; ++w) tot += sb[w];
  float mean = tot * (1.0f / D);

  float d = v - mean;
  float sq = red32_sum(d * d);
  if (lane == 0) sb2[wave] = sq;
  __syncthreads();
  float vtot = 0.0f;
#pragma unroll
  for (int w = 0; w < 8; ++w) vtot += sb2[w];
  float rstd = rsqrtf(vtot * (1.0f / D) + 1e-5f);

  out[(size_t)row * D + threadIdx.x] = d * rstd * g[threadIdx.x] + b[threadIdx.x];
}

// ---------------------------------------------------------------------------
// Host-side orchestration
// ---------------------------------------------------------------------------
extern "C" void kernel_launch(void* const* d_in, const int* in_sizes, int n_in,
                              void* d_out, int out_size, void* d_ws, size_t ws_size,
                              hipStream_t stream) {
  const float* x    = (const float*)d_in[0];   // [2048,512]
  const float* W1   = (const float*)d_in[1];   // [1024,512]
  const float* b1   = (const float*)d_in[2];   // [1024]
  const float* Wqkv = (const float*)d_in[3];   // [4,3072,1024]
  const float* bqkv = (const float*)d_in[4];   // [4,3072]
  const float* Wo   = (const float*)d_in[5];   // [4,1024,1024]
  const float* bo   = (const float*)d_in[6];   // [4,1024]
  const float* ln_g = (const float*)d_in[7];   // [4,1024]
  const float* ln_b = (const float*)d_in[8];   // [4,1024]
  const float* W2   = (const float*)d_in[9];   // [256,1024]
  const float* b2   = (const float*)d_in[10];  // [256]
  const float* gOut = (const float*)d_in[11];  // [256]
  const float* bOut = (const float*)d_in[12];  // [256]
  float* out = (float*)d_out;                  // [2048,256]

  // workspace carve-up (256B aligned)
  char* wp = (char*)d_ws;
  auto carve = [&](size_t bytes) -> void* {
    void* p = (void*)wp;
    wp += (bytes + 255) & ~(size_t)255;
    return p;
  };
  bf16_t* xB    = (bf16_t*)carve((size_t)N_TOK * D_IN * 2);
  bf16_t* w1B   = (bf16_t*)carve((size_t)HDIM * D_IN * 2);
  bf16_t* wqkvB = (bf16_t*)carve((size_t)NLAY * 3 * HDIM * HDIM * 2);
  bf16_t* woB   = (bf16_t*)carve((size_t)NLAY * HDIM * HDIM * 2);
  bf16_t* w2B   = (bf16_t*)carve((size_t)D_OUT * HDIM * 2);
  float*  hF    = (float*) carve((size_t)N_TOK * HDIM * 4);
  bf16_t* hB    = (bf16_t*)carve((size_t)N_TOK * HDIM * 2);
  bf16_t* qkvB  = (bf16_t*)carve((size_t)N_TOK * 3 * HDIM * 2);
  bf16_t* attnB = (bf16_t*)carve((size_t)N_TOK * HDIM * 2);
  float*  projF = (float*) carve((size_t)N_TOK * HDIM * 4);
  float*  yF    = (float*) carve((size_t)N_TOK * D_OUT * 4);

  auto cvt = [&](const float* src, bf16_t* dst, int n) {
    cvt_f32_bf16_kernel<<<(n + 255) / 256, 256, 0, stream>>>(src, dst, n);
  };
  cvt(x, xB, N_TOK * D_IN);
  cvt(W1, w1B, HDIM * D_IN);
  cvt(Wqkv, wqkvB, NLAY * 3 * HDIM * HDIM);
  cvt(Wo, woB, NLAY * HDIM * HDIM);
  cvt(W2, w2B, D_OUT * HDIM);

  // fc1: h = relu(x @ W1^T + b1)
  gemm_bf16_kernel<1, 1, 1><<<dim3(HDIM / 64, N_TOK / 128), 256, 0, stream>>>(
      xB, w1B, b1, hF, hB, N_TOK, HDIM, D_IN);

  const float scale = 0.125f;  // 1/sqrt(64)
  for (int layer = 0; layer < NLAY; ++layer) {
    // qkv = h @ Wqkv^T + bqkv   (bf16 out)
    gemm_bf16_kernel<0, 0, 1><<<dim3(3 * HDIM / 64, N_TOK / 128), 256, 0, stream>>>(
        hB, wqkvB + (size_t)layer * 3 * HDIM * HDIM, bqkv + (size_t)layer * 3 * HDIM,
        nullptr, qkvB, N_TOK, 3 * HDIM, HDIM);
    // flash attention
    attn_kernel<<<dim3(NHEAD, N_TOK / 128), 256, 0, stream>>>(qkvB, attnB, scale);
    // proj = attn @ Wo^T + bo   (fp32 out)
    gemm_bf16_kernel<0, 1, 0><<<dim3(HDIM / 64, N_TOK / 128), 256, 0, stream>>>(
        attnB, woB + (size_t)layer * HDIM * HDIM, bo + (size_t)layer * HDIM,
        projF, nullptr, N_TOK, HDIM, HDIM);
    // h = LN(proj + h)
    add_ln_kernel<<<N_TOK, 256, 0, stream>>>(projF, hF, hB,
                                             ln_g + (size_t)layer * HDIM,
                                             ln_b + (size_t)layer * HDIM);
  }

  // fc2: y = relu(h @ W2^T + b2)
  gemm_bf16_kernel<1, 1, 0><<<dim3(D_OUT / 64, N_TOK / 128), 256, 0, stream>>>(
      hB, w2B, b2, yF, nullptr, N_TOK, D_OUT, HDIM);
  // out = LN(y)
  final_ln_kernel<<<N_TOK, 256, 0, stream>>>(yF, gOut, bOut, out);
}
